// KascadeReuseAttention_59657095741805
// MI455X (gfx1250) — matrix operand order repaired and checked by reading, not compiled, not run on hardware
//
#include <hip/hip_runtime.h>

// ---------------------------------------------------------------------------
// Problem constants (from reference): B=4, S=4096, D=2048, H=16, HD=128
// ---------------------------------------------------------------------------
#define BB 4
#define SS 4096
#define DD 2048
#define HH 16
#define HD 128
#define TILE 16
#define NTILES 8
#define MROWS (BB * SS)            // 16384
#define CUR_TILE ((SS - 1) / TILE) // 255

typedef __attribute__((ext_vector_type(16))) __bf16 v16bf;
typedef __attribute__((ext_vector_type(8)))  float  v8f;
typedef __attribute__((ext_vector_type(4)))  float  float4v;
typedef __attribute__((ext_vector_type(8)))  unsigned short ushort8;
typedef __attribute__((ext_vector_type(4)))  unsigned short ushortx4;
typedef __attribute__((ext_vector_type(4)))  int    v4i;

union FragBF {
    v16bf v;
    ushort8 h[2];
    unsigned short s[16];
};

__device__ __forceinline__ unsigned short f2bf(float f) {
    unsigned int u = __builtin_bit_cast(unsigned int, f);
    unsigned int r = (u + 0x7FFFu + ((u >> 16) & 1u)) >> 16;
    return (unsigned short)r;
}

// ---------------------------------------------------------------------------
// gfx1250 async global->LDS copy path (guarded; falls back to reg staging)
// Signature (probe-verified): (v4i AS1* src, v4i AS3* dst, imm offset, imm cpol)
// ---------------------------------------------------------------------------
#if defined(__HIP_DEVICE_COMPILE__) &&                                   \
    __has_builtin(__builtin_amdgcn_global_load_async_to_lds_b128) &&     \
    __has_builtin(__builtin_amdgcn_s_wait_asynccnt)
#define USE_ASYNC 1
#define AS1 __attribute__((address_space(1)))
#define AS3 __attribute__((address_space(3)))
__device__ __forceinline__ void async_cp16(const void* g, void* l) {
    __builtin_amdgcn_global_load_async_to_lds_b128((AS1 v4i*)g, (AS3 v4i*)l, 0, 0);
}
#else
#define USE_ASYNC 0
#endif

// ---------------------------------------------------------------------------
// fp32 -> bf16 elementwise (vectorized x4)
// ---------------------------------------------------------------------------
__global__ __launch_bounds__(256) void cvt_bf16_kernel(const float* __restrict__ in,
                                                       unsigned short* __restrict__ out,
                                                       size_t n4) {
    size_t i = (size_t)blockIdx.x * blockDim.x + threadIdx.x;
    if (i < n4) {
        float4v f = *(const float4v*)(in + i * 4);
        ushortx4 o;
        o.x = f2bf(f.x); o.y = f2bf(f.y); o.z = f2bf(f.z); o.w = f2bf(f.w);
        *(ushortx4*)(out + i * 4) = o;
    }
}

// ---------------------------------------------------------------------------
// fp32 (R x C) -> bf16 transposed (C x R), 32x32 LDS tiles
// ---------------------------------------------------------------------------
__global__ __launch_bounds__(256) void transpose_bf16_kernel(const float* __restrict__ in,
                                                             unsigned short* __restrict__ out,
                                                             int R, int C) {
    __shared__ unsigned short tile[32][33];
    int tx = threadIdx.x & 31;
    int ty = threadIdx.x >> 5;          // 0..7
    int bx = blockIdx.x, by = blockIdx.y;
#pragma unroll
    for (int i = 0; i < 32; i += 8) {
        int r = by * 32 + ty + i;
        int c = bx * 32 + tx;
        tile[ty + i][tx] = f2bf(in[(size_t)r * C + c]);
    }
    __syncthreads();
#pragma unroll
    for (int i = 0; i < 32; i += 8) {
        int ro = bx * 32 + ty + i;      // output row = original column
        int co = by * 32 + tx;          // output col = original row
        out[(size_t)ro * R + co] = tile[tx][ty + i];
    }
}

// ---------------------------------------------------------------------------
// GEMM: C[M,N] = A[M,K] (bf16 row-major) x Bt[N,K] (bf16, B pre-transposed)
// 8 waves / block in a 4x2 grid: wave tile 32x64 (2 A-frags x 4 B-frags).
// K staged in 32-deep LDS chunks; async-LDS ping-pong (branch-free steady
// state, last iteration peeled) when available, register double-buffering
// otherwise.
// out_mode: 0 = fp32 row-major, 1 = bf16 row-major, 2 = bf16 transposed [N][M]
// ---------------------------------------------------------------------------
__global__ __launch_bounds__(256) void gemm_bf16_kernel(const unsigned short* __restrict__ A,
                                                        const unsigned short* __restrict__ Bt,
                                                        void* __restrict__ C,
                                                        int M, int N, int K, int out_mode) {
#if USE_ASYNC
    __shared__ __align__(16) unsigned short As[2][128 * 32];
    __shared__ __align__(16) unsigned short Bs[2][128 * 32];
#else
    __shared__ __align__(16) unsigned short As[1][128 * 32];
    __shared__ __align__(16) unsigned short Bs[1][128 * 32];
#endif

    const int tid     = threadIdx.x;
    const int wave    = tid >> 5;        // 0..7
    const int lane    = tid & 31;
    const int waverow = wave & 3;        // M strip (32 rows)
    const int wavecol = wave >> 2;       // N strip (64 cols)
    const int row0 = blockIdx.x * 128;
    const int col0 = blockIdx.y * 128;

    const int lrow  = tid >> 1;          // 0..127
    const int lhalf = (tid & 1) * 16;    // 0 or 16

    const int akb = (lane >> 4) * 8;     // A frag K base within chunk
    const int bkb = (lane >> 4) * 16;    // unused; B frags also use akb

    const unsigned short* aptr = A  + (size_t)(row0 + lrow) * K + lhalf;
    const unsigned short* bptr = Bt + (size_t)(col0 + lrow) * K + lhalf;

    v8f acc[2][4] = {};
    const int nchunks = K / 32;

    // shared compute body: 2 A-frags x 4 B-frags -> 8 WMMAs
    auto compute = [&](const unsigned short* as, const unsigned short* bs) {
        FragBF af[2];
#pragma unroll
        for (int mi = 0; mi < 2; ++mi) {
            int m = waverow * 32 + mi * 16 + (lane & 15);
            af[mi].h[0] = *(const ushort8*)&as[m * 32 + akb];       // K = akb..+7
            af[mi].h[1] = *(const ushort8*)&as[m * 32 + akb + 16];  // K = akb+16..+23
        }
        FragBF bfr[4];
#pragma unroll
        for (int t = 0; t < 4; ++t) {
            int n = wavecol * 64 + t * 16 + (lane & 15);
            bfr[t].h[0] = *(const ushort8*)&bs[n * 32 + akb];
            bfr[t].h[1] = *(const ushort8*)&bs[n * 32 + akb + 16];
        }
#pragma unroll
        for (int mi = 0; mi < 2; ++mi)
#pragma unroll
            for (int t = 0; t < 4; ++t)
                acc[mi][t] = __builtin_amdgcn_wmma_f32_16x16x32_bf16(
                    false, af[mi].v, false, bfr[t].v, (short)0, acc[mi][t], false, false);
    };

#if USE_ASYNC
    // prologue: async-copy chunk 0 into buffer 0
    async_cp16(aptr,     &As[0][lrow * 32 + lhalf]);
    async_cp16(aptr + 8, &As[0][lrow * 32 + lhalf + 8]);
    async_cp16(bptr,     &Bs[0][lrow * 32 + lhalf]);
    async_cp16(bptr + 8, &Bs[0][lrow * 32 + lhalf + 8]);

    // steady state: branch-free issue(next) / wait / compute(cur)
    for (int c = 0; c < nchunks - 1; ++c) {
        const int buf = c & 1;
        aptr += 32; bptr += 32;
        async_cp16(aptr,     &As[buf ^ 1][lrow * 32 + lhalf]);
        async_cp16(aptr + 8, &As[buf ^ 1][lrow * 32 + lhalf + 8]);
        async_cp16(bptr,     &Bs[buf ^ 1][lrow * 32 + lhalf]);
        async_cp16(bptr + 8, &Bs[buf ^ 1][lrow * 32 + lhalf + 8]);
        __builtin_amdgcn_s_wait_asynccnt(4);   // chunk c has landed
        __syncthreads();
        compute(As[buf], Bs[buf]);
        __syncthreads();                        // buf is overwritten next iter
    }
    // epilogue: last chunk
    __builtin_amdgcn_s_wait_asynccnt(0);
    __syncthreads();
    {
        const int lbuf = (nchunks - 1) & 1;
        compute(As[lbuf], Bs[lbuf]);
    }
#else
    ushort8 ra0 = *(const ushort8*)(aptr);
    ushort8 ra1 = *(const ushort8*)(aptr + 8);
    ushort8 rb0 = *(const ushort8*)(bptr);
    ushort8 rb1 = *(const ushort8*)(bptr + 8);
    for (int c = 0; c < nchunks; ++c) {
        __syncthreads();
        *(ushort8*)&As[0][lrow * 32 + lhalf]     = ra0;
        *(ushort8*)&As[0][lrow * 32 + lhalf + 8] = ra1;
        *(ushort8*)&Bs[0][lrow * 32 + lhalf]     = rb0;
        *(ushort8*)&Bs[0][lrow * 32 + lhalf + 8] = rb1;
        __syncthreads();
        if (c + 1 < nchunks) {  // next chunk's global loads overlap the WMMAs
            aptr += 32; bptr += 32;
            ra0 = *(const ushort8*)(aptr);
            ra1 = *(const ushort8*)(aptr + 8);
            rb0 = *(const ushort8*)(bptr);
            rb1 = *(const ushort8*)(bptr + 8);
        }
        compute(As[0], Bs[0]);
    }
#endif

    // C frag layout: lane L, VGPR r -> (m = r + 8*(L/16), n = L%16)
    const int ncol = lane & 15;
#pragma unroll
    for (int mi = 0; mi < 2; ++mi) {
#pragma unroll
        for (int t = 0; t < 4; ++t) {
#pragma unroll
            for (int r = 0; r < 8; ++r) {
                int gm = row0 + waverow * 32 + mi * 16 + (lane >> 4) * 8 + r;
                int gn = col0 + wavecol * 64 + t * 16 + ncol;
                if (out_mode == 0)      ((float*)C)[(size_t)gm * N + gn]          = acc[mi][t][r];
                else if (out_mode == 1) ((unsigned short*)C)[(size_t)gm * N + gn] = f2bf(acc[mi][t][r]);
                else                    ((unsigned short*)C)[(size_t)gn * M + gm] = f2bf(acc[mi][t][r]);
            }
        }
    }
}

// ---------------------------------------------------------------------------
// Sparse attention: 4 waves per (b, h, q-tile of 16).
// K is row-major per token; V was written transposed ([feature][token]) so
// both gathers and all WMMA fragment reads are contiguous b128 accesses.
// Gathers use the async global->LDS path when available.
// ---------------------------------------------------------------------------
__global__ __launch_bounds__(128) void attn_kernel(const unsigned short* __restrict__ Q,
                                                   const unsigned short* __restrict__ Kb,
                                                   const unsigned short* __restrict__ Vt,
                                                   const int* __restrict__ anchors,
                                                   unsigned short* __restrict__ O) {
    __shared__ __align__(16) unsigned short KVs[128][128]; // K rows, reused as Vt rows
    __shared__ __align__(16) float          Lg[16][128];   // logits / exp
    __shared__ __align__(16) unsigned short Ps[16][128];   // softmax weights bf16
    __shared__ float Lred[16][8];
    __shared__ int   tiles[8];
    __shared__ int   toks[128];

    const int qt   = blockIdx.x;   // 0..255
    const int h    = blockIdx.y;   // 0..15
    const int b    = blockIdx.z;   // 0..3
    const int tid  = threadIdx.x;  // 0..127
    const int wave = tid >> 5;     // 0..3
    const int lane = tid & 31;

    if (tid < NTILES)
        tiles[tid] = (tid == NTILES - 1) ? CUR_TILE : anchors[((b * HH) + h) * NTILES + tid];
    __syncthreads();
    toks[tid] = tiles[tid >> 4] * TILE + (tid & 15);
    __syncthreads();

    // ---- gather K rows (128 keys x 128 d, row-major) ----
    for (int i = tid; i < 128 * 16; i += 128) {
        int row = i >> 4;
        int seg = (i & 15) * 8;
        size_t g = ((size_t)(b * SS + toks[row])) * (HH * HD) + h * HD + seg;
#if USE_ASYNC
        async_cp16(&Kb[g], &KVs[row][seg]);
#else
        *(ushort8*)&KVs[row][seg] = *(const ushort8*)&Kb[g];
#endif
    }
#if USE_ASYNC
    __builtin_amdgcn_s_wait_asynccnt(0);
#endif
    __syncthreads();

    // ---- Q fragments (shared by logit waves) ----
    const int mq  = lane & 15;
    const int akb = (lane >> 4) * 8;
    const int bkb = (lane >> 4) * 16;
    const size_t qrow = ((size_t)(b * SS) + qt * 16 + mq) * (HH * HD) + h * HD;
    FragBF qf[4];
#pragma unroll
    for (int c = 0; c < 4; ++c) {
        qf[c].h[0] = *(const ushort8*)&Q[qrow + c * 32 + akb];
        qf[c].h[1] = *(const ushort8*)&Q[qrow + c * 32 + akb + 16];
    }

    // ---- logits: each wave owns 2 key tiles ----
#pragma unroll
    for (int kk = 0; kk < 2; ++kk) {
        int kt = wave * 2 + kk;
        int n  = kt * 16 + (lane & 15);
        v8f acc = {};
#pragma unroll
        for (int c = 0; c < 4; ++c) {
            FragBF bf;
            bf.h[0] = *(const ushort8*)&KVs[n][c * 32 + akb];
            bf.h[1] = *(const ushort8*)&KVs[n][c * 32 + akb + 16];
            acc = __builtin_amdgcn_wmma_f32_16x16x32_bf16(
                false, qf[c].v, false, bf.v, (short)0, acc, false, false);
        }
        int tok = toks[n];
#pragma unroll
        for (int r = 0; r < 8; ++r) {
            int m = r + 8 * (lane >> 4);
            int qpos = qt * 16 + m;
            float v = acc[r] * 0.08838834764831845f;   // 1/sqrt(128)
            if (tok > qpos) v = -1e10f;
            Lg[m][n] = v;
        }
    }
    __syncthreads();

    // ---- parallel softmax: thread owns (row = tid&15, 16-col segment = tid>>4) ----
    {
        int row = tid & 15, seg = tid >> 4;
        float mx = -3.0e38f;
#pragma unroll
        for (int j = 0; j < 16; ++j) mx = fmaxf(mx, Lg[row][seg * 16 + j]);
        Lred[row][seg] = mx;
        __syncthreads();
        float rmx = -3.0e38f;
#pragma unroll
        for (int s = 0; s < 8; ++s) rmx = fmaxf(rmx, Lred[row][s]);
        __syncthreads();
        float sum = 0.0f;
#pragma unroll
        for (int j = 0; j < 16; ++j) {
            float e = __expf(Lg[row][seg * 16 + j] - rmx);
            Lg[row][seg * 16 + j] = e;
            sum += e;
        }
        Lred[row][seg] = sum;
        __syncthreads();
        float rsum = 0.0f;
#pragma unroll
        for (int s = 0; s < 8; ++s) rsum += Lred[row][s];
        float inv = 1.0f / rsum;
#pragma unroll
        for (int j = 0; j < 16; ++j)
            Ps[row][seg * 16 + j] = f2bf(Lg[row][seg * 16 + j] * inv);
    }
    __syncthreads();

    // ---- gather V (transposed source): KVs[d][key], contiguous 16-token tiles ----
    for (int i = tid; i < 128 * 8; i += 128) {
        int d = i >> 3;
        int t = i & 7;
        size_t g = ((size_t)(h * HD + d)) * MROWS + b * SS + tiles[t] * TILE;
#if USE_ASYNC
        async_cp16(&Vt[g],     &KVs[d][t * 16]);
        async_cp16(&Vt[g + 8], &KVs[d][t * 16 + 8]);
#else
        *(ushort8*)&KVs[d][t * 16]     = *(const ushort8*)&Vt[g];
        *(ushort8*)&KVs[d][t * 16 + 8] = *(const ushort8*)&Vt[g + 8];
#endif
    }
#if USE_ASYNC
    __builtin_amdgcn_s_wait_asynccnt(0);
#endif
    __syncthreads();

    // ---- out = P (16 x 128keys) x V (128keys x 128d): each wave owns 2 d-tiles ----
    FragBF pa[4];
#pragma unroll
    for (int c = 0; c < 4; ++c) {
        pa[c].h[0] = *(const ushort8*)&Ps[mq][c * 32 + akb];
        pa[c].h[1] = *(const ushort8*)&Ps[mq][c * 32 + akb + 16];
    }
#pragma unroll
    for (int nn = 0; nn < 2; ++nn) {
        int nt = wave * 2 + nn;
        int n  = nt * 16 + (lane & 15);   // d index
        v8f acc = {};
#pragma unroll
        for (int c = 0; c < 4; ++c) {
            FragBF vf;
            vf.h[0] = *(const ushort8*)&KVs[n][c * 32 + akb];
            vf.h[1] = *(const ushort8*)&KVs[n][c * 32 + akb + 16];
            acc = __builtin_amdgcn_wmma_f32_16x16x32_bf16(
                false, pa[c].v, false, vf.v, (short)0, acc, false, false);
        }
#pragma unroll
        for (int r = 0; r < 8; ++r) {
            int m = r + 8 * (lane >> 4);
            O[((size_t)(b * SS) + qt * 16 + m) * (HH * HD) + h * HD + n] = f2bf(acc[r]);
        }
    }
}

// ---------------------------------------------------------------------------
// Host launcher
// ---------------------------------------------------------------------------
extern "C" void kernel_launch(void* const* d_in, const int* in_sizes, int n_in,
                              void* d_out, int out_size, void* d_ws, size_t ws_size,
                              hipStream_t stream) {
    (void)in_sizes; (void)n_in; (void)out_size; (void)ws_size;
    const float* x       = (const float*)d_in[0];
    const int*   anchors = (const int*)  d_in[1];
    const float* Wq      = (const float*)d_in[2];
    const float* Wk      = (const float*)d_in[3];
    const float* Wv      = (const float*)d_in[4];
    const float* Wo      = (const float*)d_in[5];
    float*       out     = (float*)d_out;

    const size_t xbytes = (size_t)MROWS * DD * 2;       // 67 MB each
    const size_t wbytes = (size_t)DD * DD * 2;          // 8 MB each

    char* w = (char*)d_ws;
    unsigned short* xb   = (unsigned short*)w; w += xbytes;
    unsigned short* wqt  = (unsigned short*)w; w += wbytes;
    unsigned short* wkt  = (unsigned short*)w; w += wbytes;
    unsigned short* wvt  = (unsigned short*)w; w += wbytes;
    unsigned short* wot  = (unsigned short*)w; w += wbytes;
    unsigned short* Qb   = (unsigned short*)w; w += xbytes;
    unsigned short* Kb   = (unsigned short*)w; w += xbytes;
    unsigned short* Vtb  = (unsigned short*)w; w += xbytes;   // transposed: [feature][token]
    unsigned short* attn = (unsigned short*)w; w += xbytes;

    // 1) precision conversion + weight transposes
    {
        size_t n4 = (size_t)MROWS * DD / 4;
        cvt_bf16_kernel<<<dim3((unsigned)(n4 / 256)), 256, 0, stream>>>(x, xb, n4);
    }
    dim3 tg(DD / 32, DD / 32);
    transpose_bf16_kernel<<<tg, 256, 0, stream>>>(Wq, wqt, DD, DD);
    transpose_bf16_kernel<<<tg, 256, 0, stream>>>(Wk, wkt, DD, DD);
    transpose_bf16_kernel<<<tg, 256, 0, stream>>>(Wv, wvt, DD, DD);
    transpose_bf16_kernel<<<tg, 256, 0, stream>>>(Wo, wot, DD, DD);

    // 2) QKV projections (bf16 WMMA); V written transposed for the attention gather
    dim3 gg(MROWS / 128, DD / 128);
    gemm_bf16_kernel<<<gg, 256, 0, stream>>>(xb, wqt, Qb,  MROWS, DD, DD, 1);
    gemm_bf16_kernel<<<gg, 256, 0, stream>>>(xb, wkt, Kb,  MROWS, DD, DD, 1);
    gemm_bf16_kernel<<<gg, 256, 0, stream>>>(xb, wvt, Vtb, MROWS, DD, DD, 2);

    // 3) sparse gathered attention
    attn_kernel<<<dim3(SS / TILE, HH, BB), 128, 0, stream>>>(Qb, Kb, Vtb, anchors, attn);

    // 4) output projection (fp32 output)
    gemm_bf16_kernel<<<gg, 256, 0, stream>>>(attn, wot, out, MROWS, DD, DD, 0);
}
